// NewAttention_25958782337611
// MI455X (gfx1250) — matrix operand order repaired and checked by reading, not compiled, not run on hardware
//
#include <hip/hip_runtime.h>

#define E       512
#define SEQ     4096
#define NBATCH  8
#define MTOT    (NBATCH * SEQ)   // 32768 rows total

#define LDS_STRIDE 80            // bytes per staged W column (64B data + 16B pad)

typedef __attribute__((ext_vector_type(16))) __bf16 v16bf;
typedef __attribute__((ext_vector_type(8)))  float  v8f;

union Frag {                 // 32 bytes: one 16-element bf16 WMMA operand per lane
    v16bf  v;
    __bf16 bf[16];
    uint4  u4[2];
};

// A fragment: two 8-element chunks (lane<16: K {k..k+7, k+16..k+23}; lane>=16: +8)
__device__ __forceinline__ void loadFragA(Frag& f, const __bf16* p, int half8) {
    f.u4[0] = *(const uint4*)(p + half8);
    f.u4[1] = *(const uint4*)(p + 16 + half8);
}

// ---------------- kernel 1: fp32 -> bf16 weight conversion --------------------
__global__ void __launch_bounds__(256)
prep_weights(const float* __restrict__ w_in, const float* __restrict__ w_out,
             __bf16* __restrict__ w_in_bf, __bf16* __restrict__ w_out_bf) {
    const int i = blockIdx.x * 256 + threadIdx.x;   // 0 .. E*E-1
    w_in_bf[i]  = (__bf16)w_in[i];
    w_out_bf[i] = (__bf16)w_out[i];
}

// ---------------- kernel 2: fp32 -> bf16 activation conversion ----------------
__global__ void __launch_bounds__(256)
conv_values(const float* __restrict__ X, __bf16* __restrict__ Xbf) {
    const size_t i = ((size_t)blockIdx.x * 256 + threadIdx.x) * 8;
    const float4 f0 = *(const float4*)(X + i);
    const float4 f1 = *(const float4*)(X + i + 4);
    Frag o;
    o.bf[0] = (__bf16)f0.x; o.bf[1] = (__bf16)f0.y;
    o.bf[2] = (__bf16)f0.z; o.bf[3] = (__bf16)f0.w;
    o.bf[4] = (__bf16)f1.x; o.bf[5] = (__bf16)f1.y;
    o.bf[6] = (__bf16)f1.z; o.bf[7] = (__bf16)f1.w;
    *(uint4*)(Xbf + i) = o.u4[0];
}

// ---------------- LDS-staged, double-buffered bf16 WMMA GEMM ------------------
// Block (8 waves): 128(M) x 128(N) tile.  Wave w: rows m0 = mb + 16*w.
// B tile W[n0..n0+127][k..k+31] staged cooperatively into LDS (8KB/step, x2 buf).
// Compute phase software-pipelines the LDS B-fragment loads one deep so each
// WMMA waits only on the *previous* fragment's (already-arrived) data.
template<bool BF_OUT>
__global__ void __launch_bounds__(256)
gemm_bf16(const __bf16* __restrict__ Abf,     // [MTOT, E] bf16
          const __bf16* __restrict__ Wbf,     // [E, E] bf16 row-major (n,k)
          void* __restrict__ OutP)            // [MTOT, E] bf16 or fp32
{
    __shared__ __align__(16) unsigned char smem[2][128 * LDS_STRIDE];  // 20 KB

    const int tid   = threadIdx.x;
    const int lane  = tid & 31;
    const int wave  = tid >> 5;
    const int mb    = (blockIdx.x >> 2) * 128;
    const int n0    = (blockIdx.x & 3) * 128;
    const int m0    = mb + wave * 16;
    const int l15   = lane & 15;
    const int half8 = (lane < 16) ? 0 : 8;    // A-fragment K sub-chunk select
    const int boff  = (lane < 16) ? 0 : 16;   // B-fragment K half byte offset in LDS

    // staging role: each thread copies 32B (16 bf16 of one W column) per k-step
    const int scol  = tid >> 1;               // 0..127 : column within N tile
    const int shalf = tid & 1;                // 0/1    : k sub-chunk of 16
    const __bf16* wsrc = Wbf + (size_t)(n0 + scol) * E + shalf * 16;
    const int     sdst = scol * LDS_STRIDE + shalf * 32;

    const v8f zero = {0.f, 0.f, 0.f, 0.f, 0.f, 0.f, 0.f, 0.f};
    v8f acc[8];
#pragma unroll
    for (int i = 0; i < 8; ++i) acc[i] = zero;

    const __bf16* arow = Abf + (size_t)(m0 + l15) * E;
    // per-wave LDS read base for B fragments (fragment nf at +nf*16*LDS_STRIDE)
    const unsigned char* bbase = &smem[0][l15 * LDS_STRIDE + boff];
    const int bufBytes = 128 * LDS_STRIDE;

    // ---- prologue: stage k = 0 into smem[0]; load first A fragment ----
    {
        const uint4 s0 = *(const uint4*)(wsrc);
        const uint4 s1 = *(const uint4*)(wsrc + 8);
        *(uint4*)(&smem[0][sdst])      = s0;
        *(uint4*)(&smem[0][sdst + 16]) = s1;
    }
    Frag aCur;
    loadFragA(aCur, arow, half8);
    __syncthreads();

    for (int i = 0; i < 16; ++i) {
        const int kc  = i * 32;
        const int kn  = (i < 15) ? kc + 32 : kc;   // clamped prefetch (discarded)
        const int cur = i & 1, nxt = cur ^ 1;

        // issue next-step global loads (overlap with compute below)
        const uint4 t0 = *(const uint4*)(wsrc + kn);
        const uint4 t1 = *(const uint4*)(wsrc + kn + 8);
        Frag aNext;
        loadFragA(aNext, arow + kn, half8);

        // compute from current LDS buffer, B-fragment loads pipelined one deep
        const unsigned char* bp = bbase + cur * bufBytes;
        Frag bCur, bNxt;
        bCur.u4[0] = *(const uint4*)(bp);
        bCur.u4[1] = *(const uint4*)(bp + 32);
#pragma unroll
        for (int nf = 0; nf < 8; ++nf) {
            if (nf < 7) {
                const unsigned char* pn = bp + (nf + 1) * (16 * LDS_STRIDE);
                bNxt.u4[0] = *(const uint4*)(pn);
                bNxt.u4[1] = *(const uint4*)(pn + 32);
            }
            acc[nf] = __builtin_amdgcn_wmma_f32_16x16x32_bf16(
                false, aCur.v, false, bCur.v, (short)0, acc[nf], false, false);
            bCur = bNxt;   // fully unrolled: becomes register renaming, no copies
        }

        // commit staged data to the other LDS buffer
        *(uint4*)(&smem[nxt][sdst])      = t0;
        *(uint4*)(&smem[nxt][sdst + 16]) = t1;
        aCur = aNext;
        __syncthreads();
    }

    // C/D layout: VGPR i -> row m0 + i + (lane<16 ? 0 : 8), col n0 + nf*16 + (lane&15)
#pragma unroll
    for (int nf = 0; nf < 8; ++nf) {
        const int ncol = n0 + nf * 16 + l15;
#pragma unroll
        for (int i = 0; i < 8; ++i) {
            const size_t off = (size_t)(m0 + half8 + i) * E + ncol;
            if constexpr (BF_OUT) ((__bf16*)OutP)[off] = (__bf16)acc[nf][i];
            else                  ((float*)OutP)[off]  = acc[nf][i];
        }
    }
}

// ---------------- per-head shifted 5-tap Gaussian filter ----------------------
__global__ void __launch_bounds__(256)
filter_k(const __bf16* __restrict__ Vin,   // [MTOT, E] bf16
         __bf16* __restrict__ Fout)        // [MTOT, E] bf16
{
    const int idx = blockIdx.x * 256 + threadIdx.x;   // one 8-wide chunk each
    const int c   = (idx & 63) * 8;                   // column base 0..504
    const int g   = idx >> 6;                         // global row
    const int b   = g >> 12;                          // batch
    const int s   = g & (SEQ - 1);                    // seq position
    const int h3  = (c >> 6) & 3;                     // head pattern {0,-1,+1,0}
    const int sh  = (h3 == 1) ? -1 : ((h3 == 2) ? 1 : 0);

    float acc[8] = {0.f, 0.f, 0.f, 0.f, 0.f, 0.f, 0.f, 0.f};
    const float filt[5] = {0.05399096651f, 0.24197072451f, 0.39894228040f,
                           0.24197072451f, 0.05399096651f};
#pragma unroll
    for (int j = 0; j < 5; ++j) {
        const int r = s + sh + j - 2;
        if (r >= 0 && r < SEQ) {
            Frag d;
            d.u4[0] = *(const uint4*)(Vin + ((size_t)(b * SEQ + r) * E + c));
            const float w = filt[j];
#pragma unroll
            for (int t = 0; t < 8; ++t) acc[t] += w * (float)d.bf[t];
        }
    }
    Frag o;
#pragma unroll
    for (int t = 0; t < 8; ++t) o.bf[t] = (__bf16)acc[t];
    *(uint4*)(Fout + ((size_t)g * E + c)) = o.u4[0];
}

// -----------------------------------------------------------------------------
extern "C" void kernel_launch(void* const* d_in, const int* in_sizes, int n_in,
                              void* d_out, int out_size, void* d_ws, size_t ws_size,
                              hipStream_t stream) {
    const float* values = (const float*)d_in[0];
    // d_in[1] = keys, d_in[2] = queries: unused by the reference attention path
    const float* w_in   = (const float*)d_in[3];
    const float* w_out  = (const float*)d_in[4];

    // workspace layout (bf16 elements)
    __bf16* w_in_bf  = (__bf16*)d_ws;                        // 512 KB
    __bf16* w_out_bf = w_in_bf  + (size_t)E * E;             // 512 KB
    __bf16* x_bf     = w_out_bf + (size_t)E * E;             // 32 MB (values bf16)
    __bf16* v_bf     = x_bf     + (size_t)MTOT * E;          // 32 MB (projected v)
    __bf16* f_bf     = v_bf     + (size_t)MTOT * E;          // 32 MB (filtered)

    const int gemm_blocks = (MTOT / 128) * (E / 128);        // 1024
    const int elem_blocks = (int)(((size_t)MTOT * E / 8) / 256);  // 8192

    prep_weights<<<(E * E) / 256, 256, 0, stream>>>(w_in, w_out, w_in_bf, w_out_bf);
    conv_values<<<elem_blocks, 256, 0, stream>>>(values, x_bf);
    gemm_bf16<true><<<gemm_blocks, 256, 0, stream>>>(x_bf, w_in_bf, v_bf);
    filter_k<<<elem_blocks, 256, 0, stream>>>(v_bf, f_bf);
    gemm_bf16<false><<<gemm_blocks, 256, 0, stream>>>(f_bf, w_out_bf, (float*)d_out);
}
